// QunatEncoderBlock_53644141527091
// MI455X (gfx1250) — compile-verified, matrix operand order checked by reading, not run on hardware
//
#include <hip/hip_runtime.h>
#include <hip/hip_bf16.h>
#include <stdint.h>

typedef int v8i __attribute__((ext_vector_type(8)));
typedef int vi4 __attribute__((vector_size(16)));

#define DIM   768
#define NH    12
#define HD    64
#define MLPD  3072
#define WS_   14
#define NTOK  196     // WS*WS
#define NPAD  208     // 13*16
#define NWIN  100     // B * 5 * 5
#define MQ    19600   // NWIN * NTOK  (divisible by 16)
#define MXROW 16384   // B*H*W
#define HEADS 1200    // NWIN * NH

// ---------------------------------------------------------------- helpers

__device__ __forceinline__ int laneid() { return threadIdx.x & 31; }

__device__ __forceinline__ v8i v8zero() {
  v8i z;
#pragma unroll
  for (int i = 0; i < 8; ++i) z[i] = 0;
  return z;
}

__device__ __forceinline__ v8i wmma_iu8(v8i a, v8i b, v8i c) {
  // signed A, signed B, i32 accumulate  -> v_wmma_i32_16x16x64_iu8
  return __builtin_amdgcn_wmma_i32_16x16x64_iu8(true, a, true, b, c, false, false);
}

// ---- CDNA5 async global->LDS path (ASYNCcnt), with safe fallback ---------

#if __has_builtin(__builtin_amdgcn_global_load_async_to_lds_b128)
#define HAVE_ASYNC_LDS 1
typedef __attribute__((address_space(1))) vi4 gvi4;   // global 16B vector
typedef __attribute__((address_space(3))) vi4 lvi4;   // LDS 16B vector
#else
#define HAVE_ASYNC_LDS 0
#endif

// copy 16 bytes from global to LDS (per-lane addresses)
__device__ __forceinline__ void cp16_g2l(const int8_t* g, int8_t* l) {
#if HAVE_ASYNC_LDS
  __builtin_amdgcn_global_load_async_to_lds_b128((gvi4*)g, (lvi4*)l, 0, 0);
#else
  *(int4*)l = *(const int4*)g;
#endif
}

__device__ __forceinline__ void async_wait() {
#if HAVE_ASYNC_LDS
#if __has_builtin(__builtin_amdgcn_s_wait_asynccnt)
  __builtin_amdgcn_s_wait_asynccnt(0);
#else
  asm volatile("s_wait_asynccnt 0x0" ::: "memory");
#endif
#endif
}

// Load a 16x64 int8 operand tile (A or B layout for 16x16x64 IU8 WMMA).
// ISA layout: lane m (0..15) holds row m bytes K in {0-7,16-23,32-39,48-55},
// lane m+16 holds K in {8-15,24-31,40-47,56-63}.
template <typename T>
__device__ __forceinline__ v8i ld_frag(const T* basep, int stride) {
  const int lane = laneid();
  const int8_t* p = (const int8_t*)basep + (lane & 15) * stride + ((lane >> 4) & 1) * 8;
  int2 d0 = *(const int2*)(p + 0);
  int2 d1 = *(const int2*)(p + 16);
  int2 d2 = *(const int2*)(p + 32);
  int2 d3 = *(const int2*)(p + 48);
  v8i r;
  r[0] = d0.x; r[1] = d0.y; r[2] = d1.x; r[3] = d1.y;
  r[4] = d2.x; r[5] = d2.y; r[6] = d3.x; r[7] = d3.y;
  return r;
}

__device__ __forceinline__ int8_t fq8(float v, float s) {
  float q = rintf(v / s);                 // round-half-even, matches jnp.round
  q = fminf(fmaxf(q, -128.f), 127.f);
  return (int8_t)q;
}

// ---------------------------------------------------------------- weight quant

__global__ __launch_bounds__(256) void quant_w_kernel(const float* __restrict__ w,
                                                      int8_t* __restrict__ o, int n,
                                                      const float* __restrict__ wsc, int wi) {
  int i = blockIdx.x * 256 + threadIdx.x;
  if (i < n) o[i] = fq8(w[i], wsc[wi]);
}

// ---------------------------------------------------------------- layernorm + quant
template <int WINDOWED>
__global__ __launch_bounds__(256) void ln_quant_kernel(const float* __restrict__ x,
                                                       const float* __restrict__ g,
                                                       const float* __restrict__ b,
                                                       const float* __restrict__ as, int asi,
                                                       int8_t* __restrict__ out) {
  const int m = blockIdx.x;
  const int tid = threadIdx.x;
  int8_t* orow = out + (size_t)m * DIM;
  const float* src;
  if (WINDOWED) {
    int w = m / NTOK, t = m - w * NTOK;
    int bb = w / 25, wh = (w % 25) / 5, ww = w % 5;
    int i = t / WS_, j = t - (t / WS_) * WS_;
    int h = wh * WS_ + i, gw = ww * WS_ + j;
    if (h >= 64 || gw >= 64) {        // padded token: LN(pad)=0 -> quant -> 0
      for (int c = tid; c < DIM; c += 256) orow[c] = 0;
      return;
    }
    src = x + (size_t)((bb * 64 + h) * 64 + gw) * DIM;
  } else {
    src = x + (size_t)m * DIM;
  }
  float v0 = src[tid], v1 = src[tid + 256], v2 = src[tid + 512];
  __shared__ float rs[256], rq[256];
  rs[tid] = v0 + v1 + v2;
  rq[tid] = v0 * v0 + v1 * v1 + v2 * v2;
  __syncthreads();
  for (int off = 128; off > 0; off >>= 1) {
    if (tid < off) { rs[tid] += rs[tid + off]; rq[tid] += rq[tid + off]; }
    __syncthreads();
  }
  float mean = rs[0] * (1.0f / DIM);
  float var  = rq[0] * (1.0f / DIM) - mean * mean;
  float rstd = rsqrtf(var + 1e-6f);
  float s = as[asi];
  int c = tid;        orow[c] = fq8((v0 - mean) * rstd * g[c] + b[c], s);
  c = tid + 256;      orow[c] = fq8((v1 - mean) * rstd * g[c] + b[c], s);
  c = tid + 512;      orow[c] = fq8((v2 - mean) * rstd * g[c] + b[c], s);
}

// ---------------------------------------------------------------- int8 GEMM (C = A * Bw^T)

struct GemmEpi {
  const float* as;  const float* wsc;  const float* bias;
  int8_t* q8; int8_t* k8; int8_t* v8p;   // MODE 0 (qkv) outputs
  const float* fin; float* fout;         // MODE 1 (proj+res) / MODE 3 (lin2+res)
  int8_t* o8;                            // MODE 2 (lin1 -> gelu -> int8)
};

template <int MODE>
__global__ __launch_bounds__(256) void gemm_i8_kernel(const int8_t* __restrict__ A,
                                                      const int8_t* __restrict__ Bw,
                                                      int M, int N, int K, GemmEpi e) {
  // double-buffered tiles, 80-byte row stride (conflict-free ds_load_b64 frags)
  __shared__ __align__(16) int8_t lA[2][128 * 80];
  __shared__ __align__(16) int8_t lB[2][128 * 80];
  const int tid  = threadIdx.x;
  const int lane = tid & 31;
  const int wave = tid >> 5;
  const int wm = wave >> 1;       // 0..3 : M sub-block
  const int wn = wave & 1;        // 0..1 : N sub-block
  const int m0 = blockIdx.x * 128;
  const int n0 = blockIdx.y * 128;

  v8i acc[2][4];
#pragma unroll
  for (int i = 0; i < 2; ++i)
#pragma unroll
    for (int j = 0; j < 4; ++j) acc[i][j] = v8zero();

  const int lrow  = tid >> 1;
  const int lhalf = (tid & 1) * 32;

  // stage one 128x64 A tile + 128x64 B tile into buffer `buf` (async path)
  auto stage = [&](int k0, int buf) {
    {
      int gr = m0 + lrow;
      int8_t* d = &lA[buf][lrow * 80 + lhalf];
      if (gr < M) {
        const int8_t* s = A + (size_t)gr * K + k0 + lhalf;
        cp16_g2l(s, d);
        cp16_g2l(s + 16, d + 16);
      } else {
        *(int4*)d = make_int4(0, 0, 0, 0);
        *(int4*)(d + 16) = make_int4(0, 0, 0, 0);
      }
    }
    {
      int gr = n0 + lrow;   // N is always a multiple of 128
      const int8_t* s = Bw + (size_t)gr * K + k0 + lhalf;
      int8_t* d = &lB[buf][lrow * 80 + lhalf];
      cp16_g2l(s, d);
      cp16_g2l(s + 16, d + 16);
    }
  };

  const int nk = K >> 6;
  stage(0, 0);
  for (int ik = 0; ik < nk; ++ik) {
    const int buf = ik & 1;
    async_wait();           // this wave's async fills done
    __syncthreads();        // everyone's fills done -> tile readable
    if (ik + 1 < nk) stage((ik + 1) << 6, buf ^ 1);   // prefetch next tile
    v8i af[2], bf[4];
#pragma unroll
    for (int i = 0; i < 2; ++i) af[i] = ld_frag(&lA[buf][(wm * 32 + i * 16) * 80], 80);
#pragma unroll
    for (int j = 0; j < 4; ++j) bf[j] = ld_frag(&lB[buf][(wn * 64 + j * 16) * 80], 80);
#pragma unroll
    for (int i = 0; i < 2; ++i)
#pragma unroll
      for (int j = 0; j < 4; ++j) acc[i][j] = wmma_iu8(af[i], bf[j], acc[i][j]);
    __syncthreads();        // tile consumed -> may be overwritten at ik+2
  }

  float sAB;
  if (MODE == 0)      sAB = e.as[4] * e.wsc[0];
  else if (MODE == 1) sAB = e.as[5] * e.wsc[1];
  else if (MODE == 2) sAB = e.as[6] * e.wsc[2];
  else                sAB = e.as[7] * e.wsc[3];

  const int rbase = ((lane >> 4) & 1) * 8;   // C layout: row = v + 8*(lane>>4), col = lane&15
  const int col   = lane & 15;
#pragma unroll
  for (int i = 0; i < 2; ++i) {
    int mrow0 = m0 + wm * 32 + i * 16;
#pragma unroll
    for (int j = 0; j < 4; ++j) {
      int ncol = n0 + wn * 64 + j * 16 + col;
      float bias = e.bias[ncol];
#pragma unroll
      for (int v = 0; v < 8; ++v) {
        int m = mrow0 + rbase + v;
        if (m >= M) continue;
        float val = (float)acc[i][j][v] * sAB + bias;
        if (MODE == 0) {
          int part = ncol / DIM;            // 0=q 1=k 2=v
          int rem  = ncol - part * DIM;
          int head = rem >> 6, c = rem & 63;
          int w = m / NTOK, t = m - w * NTOK;
          size_t idx = ((size_t)(w * NH + head) * NPAD + t) * HD + c;
          int8_t q = fq8(val, e.as[part]);
          if (part == 0) e.q8[idx] = q;
          else if (part == 1) e.k8[idx] = q;
          else e.v8p[idx] = q;
        } else if (MODE == 1) {             // proj: un-window, crop, + shortcut
          int w = m / NTOK, t = m - w * NTOK;
          int bb = w / 25, wh = (w % 25) / 5, ww = w % 5;
          int ii = t / WS_, jj = t - ii * WS_;
          int h = wh * WS_ + ii, gw = ww * WS_ + jj;
          if (h < 64 && gw < 64) {
            size_t idx = (size_t)((bb * 64 + h) * 64 + gw) * DIM + ncol;
            e.fout[idx] = e.fin[idx] + val;
          }
        } else if (MODE == 2) {             // lin1: exact gelu + requant(a7)
          float gl = 0.5f * val * (1.0f + erff(val * 0.70710678118654752f));
          e.o8[(size_t)m * MLPD + ncol] = fq8(gl, e.as[7]);
        } else {                            // lin2: + residual -> final out
          size_t idx = (size_t)m * DIM + ncol;
          e.fout[idx] = e.fin[idx] + val;
        }
      }
    }
  }
}

// ---------------------------------------------------------------- rel-pos bias precompute
__global__ __launch_bounds__(32) void relbias_kernel(const int8_t* __restrict__ qb,
                                                     const float* __restrict__ rph,
                                                     const float* __restrict__ rpw,
                                                     const float* __restrict__ as,
                                                     float* __restrict__ bH,
                                                     float* __restrict__ bW) {
  int blk  = blockIdx.x;              // head*NTOK + t
  int head = blk / NTOK;
  int t    = blk - head * NTOK;
  int lane = threadIdx.x;
  __shared__ int8_t qrow[64];
  if (lane < 16)
    ((int*)qrow)[lane] = ((const int*)(qb + ((size_t)head * NPAD + t) * HD))[lane];
  __syncthreads();
  float a0 = as[0];
  int i = t / WS_, j = t - (t / WS_) * WS_;
  if (lane < 28) {
    bool isH = lane < 14;
    int k = isH ? lane : lane - 14;
    int ridx = (isH ? i : j) - k + (WS_ - 1);
    const float* R = (isH ? rph : rpw) + (size_t)ridx * HD;
    float acc = 0.f;
#pragma unroll
    for (int c = 0; c < HD; ++c) acc += (float)qrow[c] * R[c];
    float outv = a0 * acc;
    if (isH) bH[(size_t)blk * WS_ + k] = outv;
    else     bW[(size_t)blk * WS_ + k] = outv;
  }
}

// ---------------------------------------------------------------- attention core
__global__ __launch_bounds__(32) void attn_kernel(const int8_t* __restrict__ qb,
                                                  const int8_t* __restrict__ kb,
                                                  const int8_t* __restrict__ vb,
                                                  const float* __restrict__ bH,
                                                  const float* __restrict__ bW,
                                                  const float* __restrict__ as,
                                                  int8_t* __restrict__ attO) {
  __shared__ __align__(16) int8_t sK[NPAD * 80];    // K rows, row-major (B operand)
  __shared__ __align__(16) int8_t sVt[HD * 272];    // V transposed: [col][K], K padded->256
  __shared__ __align__(16) int8_t sP[16 * 272];     // quantized probs (A operand), K->256
  int blk  = blockIdx.x;
  int head = blk / 13;
  int qt   = blk - head * 13;
  int lane = threadIdx.x;

  // async-stage K rows into LDS (4 x b128 per row)
  for (int r = lane; r < NPAD; r += 32) {
    const int8_t* s = kb + ((size_t)head * NPAD + r) * HD;
    int8_t* d = sK + r * 80;
    cp16_g2l(s, d); cp16_g2l(s + 16, d + 16);
    cp16_g2l(s + 32, d + 32); cp16_g2l(s + 48, d + 48);
  }
  for (int i = lane; i < HD * 272 / 4; i += 32) ((int*)sVt)[i] = 0;
  for (int i = lane; i < 16 * 272 / 4; i += 32) ((int*)sP)[i] = 0;
  __syncthreads();
  for (int i = lane; i < NPAD * 16; i += 32) {      // scatter V -> V^T
    int k = i >> 4, cw = i & 15;
    unsigned wv = ((const unsigned*)(vb + ((size_t)head * NPAD + k) * HD))[cw];
    int cb = cw * 4;
    sVt[(cb + 0) * 272 + k] = (int8_t)(wv & 0xff);
    sVt[(cb + 1) * 272 + k] = (int8_t)((wv >> 8) & 0xff);
    sVt[(cb + 2) * 272 + k] = (int8_t)((wv >> 16) & 0xff);
    sVt[(cb + 3) * 272 + k] = (int8_t)((wv >> 24) & 0xff);
  }
  v8i qf = ld_frag(qb + ((size_t)head * NPAD + qt * 16) * HD, HD);
  async_wait();
  __syncthreads();

  v8i sacc[13];
#pragma unroll
  for (int nt = 0; nt < 13; ++nt) {
    v8i bf = ld_frag(sK + nt * 16 * 80, 80);
    sacc[nt] = wmma_iu8(qf, bf, v8zero());
  }

  const float sqk = as[0] * as[1] * 0.125f;   // a0*a1*HD^-0.5
  const float a3  = as[3];
  const int rbase = ((lane >> 4) & 1) * 8;
  const int col   = lane & 15;

#pragma unroll
  for (int v = 0; v < 8; ++v) {
    int rloc = rbase + v;
    int m = qt * 16 + rloc;
    int msafe = (m < NTOK) ? m : 0;
    const float* bHrow = bH + ((size_t)head * NTOK + msafe) * WS_;
    const float* bWrow = bW + ((size_t)head * NTOK + msafe) * WS_;
    float sv[13];
    float mx = -3.0e38f;
#pragma unroll
    for (int nt = 0; nt < 13; ++nt) {
      int n = nt * 16 + col;
      float xv;
      if (m < NTOK && n < NTOK)
        xv = (float)sacc[nt][v] * sqk + bHrow[n / WS_] + bWrow[n % WS_];
      else
        xv = -1.0e30f;
      sv[nt] = xv;
      mx = fmaxf(mx, xv);
    }
    for (int d = 8; d >= 1; d >>= 1) mx = fmaxf(mx, __shfl_xor(mx, d, 32));
    float sum = 0.f;
#pragma unroll
    for (int nt = 0; nt < 13; ++nt) { sv[nt] = expf(sv[nt] - mx); sum += sv[nt]; }
    for (int d = 8; d >= 1; d >>= 1) sum += __shfl_xor(sum, d, 32);
    float inv = 1.0f / sum;
#pragma unroll
    for (int nt = 0; nt < 13; ++nt) {
      int n = nt * 16 + col;
      sP[rloc * 272 + n] = fq8(sv[nt] * inv, a3);
    }
  }
  __syncthreads();

  v8i oacc[4];
#pragma unroll
  for (int j = 0; j < 4; ++j) oacc[j] = v8zero();
#pragma unroll
  for (int ks = 0; ks < 4; ++ks) {
    v8i af = ld_frag(sP + ks * 64, 272);
#pragma unroll
    for (int j = 0; j < 4; ++j) {
      v8i bf = ld_frag(sVt + (j * 16) * 272 + ks * 64, 272);
      oacc[j] = wmma_iu8(af, bf, oacc[j]);
    }
  }

  const float spv = as[3] * as[2];
  const float a5  = as[5];
  int w = head / NH, hd = head - w * NH;
#pragma unroll
  for (int j = 0; j < 4; ++j) {
#pragma unroll
    for (int v = 0; v < 8; ++v) {
      int t = qt * 16 + rbase + v;
      if (t < NTOK) {
        int c = hd * HD + j * 16 + col;
        attO[(size_t)(w * NTOK + t) * DIM + c] = fq8((float)oacc[j][v] * spv, a5);
      }
    }
  }
}

// ---------------------------------------------------------------- launch

extern "C" void kernel_launch(void* const* d_in, const int* in_sizes, int n_in,
                              void* d_out, int out_size, void* d_ws, size_t ws_size,
                              hipStream_t stream) {
  const float* x     = (const float*)d_in[0];
  const float* ln1w  = (const float*)d_in[1];
  const float* ln1b  = (const float*)d_in[2];
  const float* ln2w  = (const float*)d_in[3];
  const float* ln2b  = (const float*)d_in[4];
  const float* qkvw  = (const float*)d_in[5];
  const float* qkvb  = (const float*)d_in[6];
  const float* projw = (const float*)d_in[7];
  const float* projb = (const float*)d_in[8];
  const float* lin1w = (const float*)d_in[9];
  const float* lin1b = (const float*)d_in[10];
  const float* lin2w = (const float*)d_in[11];
  const float* lin2b = (const float*)d_in[12];
  const float* rph   = (const float*)d_in[13];
  const float* rpw   = (const float*)d_in[14];
  const float* as    = (const float*)d_in[15];
  const float* wsc   = (const float*)d_in[16];
  float* outp = (float*)d_out;

  char* base = (char*)d_ws;
  size_t off = 0;
  auto alloc = [&](size_t b) -> char* {
    char* p = base + off;
    off = (off + b + 255) & ~(size_t)255;
    return p;
  };
  int8_t* wq_qkv  = (int8_t*)alloc((size_t)3 * DIM * DIM);
  int8_t* wq_proj = (int8_t*)alloc((size_t)DIM * DIM);
  int8_t* wq_lin1 = (int8_t*)alloc((size_t)MLPD * DIM);
  int8_t* wq_lin2 = (int8_t*)alloc((size_t)DIM * MLPD);
  int8_t* A1      = (int8_t*)alloc((size_t)MQ * DIM);
  int8_t* qb      = (int8_t*)alloc((size_t)HEADS * NPAD * HD);
  int8_t* kb      = (int8_t*)alloc((size_t)HEADS * NPAD * HD);
  int8_t* vb      = (int8_t*)alloc((size_t)HEADS * NPAD * HD);
  float*  bH      = (float*)alloc((size_t)HEADS * NTOK * WS_ * 4);
  float*  bW      = (float*)alloc((size_t)HEADS * NTOK * WS_ * 4);
  int8_t* attO    = (int8_t*)alloc((size_t)MQ * DIM);
  float*  x2      = (float*)alloc((size_t)MXROW * DIM * 4);
  int8_t* y2      = (int8_t*)alloc((size_t)MXROW * DIM);
  int8_t* hb      = (int8_t*)alloc((size_t)MXROW * MLPD);

  // 1) quantize weights
  auto wq = [&](const float* w, int8_t* o, int n, int wi) {
    quant_w_kernel<<<(n + 255) / 256, 256, 0, stream>>>(w, o, n, wsc, wi);
  };
  wq(qkvw,  wq_qkv,  3 * DIM * DIM, 0);
  wq(projw, wq_proj, DIM * DIM,     1);
  wq(lin1w, wq_lin1, MLPD * DIM,    2);
  wq(lin2w, wq_lin2, DIM * MLPD,    3);

  // 2) LN1 + window partition + quant(a4)
  ln_quant_kernel<1><<<MQ, 256, 0, stream>>>(x, ln1w, ln1b, as, 4, A1);

  // 3) zero q/k/v pad rows, qkv GEMM with fused per-head requant
  (void)hipMemsetAsync(qb, 0, (size_t)HEADS * NPAD * HD, stream);
  (void)hipMemsetAsync(kb, 0, (size_t)HEADS * NPAD * HD, stream);
  (void)hipMemsetAsync(vb, 0, (size_t)HEADS * NPAD * HD, stream);
  {
    GemmEpi e{}; e.as = as; e.wsc = wsc; e.bias = qkvb; e.q8 = qb; e.k8 = kb; e.v8p = vb;
    gemm_i8_kernel<0><<<dim3((MQ + 127) / 128, (3 * DIM) / 128), 256, 0, stream>>>(
        A1, wq_qkv, MQ, 3 * DIM, DIM, e);
  }

  // 4) rel-pos bias
  relbias_kernel<<<HEADS * NTOK, 32, 0, stream>>>(qb, rph, rpw, as, bH, bW);

  // 5) attention
  attn_kernel<<<HEADS * 13, 32, 0, stream>>>(qb, kb, vb, bH, bW, as, attO);

  // 6) proj GEMM + un-window + residual -> x2
  {
    GemmEpi e{}; e.as = as; e.wsc = wsc; e.bias = projb; e.fin = x; e.fout = x2;
    gemm_i8_kernel<1><<<dim3((MQ + 127) / 128, DIM / 128), 256, 0, stream>>>(
        attO, wq_proj, MQ, DIM, DIM, e);
  }

  // 7) LN2 + quant(a6)
  ln_quant_kernel<0><<<MXROW, 256, 0, stream>>>(x2, ln2w, ln2b, as, 6, y2);

  // 8) MLP lin1 (+gelu +quant a7)
  {
    GemmEpi e{}; e.as = as; e.wsc = wsc; e.bias = lin1b; e.o8 = hb;
    gemm_i8_kernel<2><<<dim3(MXROW / 128, MLPD / 128), 256, 0, stream>>>(
        y2, wq_lin1, MXROW, MLPD, DIM, e);
  }

  // 9) MLP lin2 + residual -> d_out
  {
    GemmEpi e{}; e.as = as; e.wsc = wsc; e.bias = lin2b; e.fin = x2; e.fout = outp;
    gemm_i8_kernel<3><<<dim3(MXROW / 128, DIM / 128), 256, 0, stream>>>(
        hb, wq_lin2, MXROW, DIM, MLPD, e);
  }
}
